// BertAttention_35433480192576
// MI455X (gfx1250) — matrix operand order repaired and checked by reading, not compiled, not run on hardware
//
#include <hip/hip_runtime.h>
#include <hip/hip_bf16.h>

// ---------------------------------------------------------------------------
// Tree-Transformer BERT attention layer for MI455X (gfx1250), wave32 + WMMA.
// B=4, S=1024, H=1024, NH=16, DH=64.
// Outputs (concat, f32): out[B,S,H], scores[B,NH,S,S], g_attn[B,S,S], break_prob[B,S,S]
// ---------------------------------------------------------------------------

#define Bb 4
#define Ss 1024
#define Hh 1024
#define NHh 16
#define DHh 64
#define Nrows 4096   // B*S

typedef __attribute__((ext_vector_type(8)))  float          v8f;
typedef __attribute__((ext_vector_type(16))) __bf16         v16bf;
typedef __attribute__((ext_vector_type(16))) unsigned short u16x16;
typedef __attribute__((ext_vector_type(8)))  unsigned short u16x8;

__device__ __forceinline__ unsigned short f2bf(float f) {
  unsigned int u = __builtin_bit_cast(unsigned int, f);
  unsigned int r = u + 0x7FFFu + ((u >> 16) & 1u);   // round-to-nearest-even
  return (unsigned short)(r >> 16);
}

__device__ __forceinline__ v8f vzero8() {
  v8f z;
#pragma unroll
  for (int i = 0; i < 8; ++i) z[i] = 0.0f;
  return z;
}

// B-fragment: 16 contiguous halves
__device__ __forceinline__ v16bf ld_frag16(const unsigned short* p) {
  u16x16 u = *reinterpret_cast<const u16x16*>(p);
  return __builtin_bit_cast(v16bf, u);
}
// A-fragment: two contiguous 8-half segments (K layout of 16-bit A matrix)
__device__ __forceinline__ v16bf ld_fragA(const unsigned short* p0, const unsigned short* p1) {
  u16x8 a = *reinterpret_cast<const u16x8*>(p0);
  u16x8 b = *reinterpret_cast<const u16x8*>(p1);
  u16x16 u;
#pragma unroll
  for (int i = 0; i < 8; ++i) { u[i] = a[i]; u[i + 8] = b[i]; }
  return __builtin_bit_cast(v16bf, u);
}

__device__ __forceinline__ v8f wmma_bf16(v16bf a, v16bf b, v8f c) {
  return __builtin_amdgcn_wmma_f32_16x16x32_bf16(false, a, false, b, (short)0, c, false, false);
}

// ---------------------------------------------------------------------------
// f32 -> bf16 conversion (grid: one element per thread)
// ---------------------------------------------------------------------------
__global__ __launch_bounds__(256) void k_cvt_bf16(const float* __restrict__ src,
                                                  unsigned short* __restrict__ dst, int n) {
  int i = blockIdx.x * 256 + threadIdx.x;
  if (i < n) dst[i] = f2bf(src[i]);
}

// ---------------------------------------------------------------------------
// Row LayerNorm over H=1024. One block (256 thr = 8 waves) per row.
// Writes f32 (out_f32 != null) or bf16.
// ---------------------------------------------------------------------------
__global__ __launch_bounds__(256) void k_layernorm(const float* __restrict__ x,
                                                   const float* __restrict__ sc,
                                                   const float* __restrict__ bi,
                                                   float* __restrict__ out_f32,
                                                   unsigned short* __restrict__ out_bf16) {
  __shared__ float red0[8], red1[8];
  const int row = blockIdx.x;
  const float* xr = x + (size_t)row * Hh;
  float s = 0.f, s2 = 0.f;
  for (int c = threadIdx.x; c < Hh; c += 256) { float v = xr[c]; s += v; s2 += v * v; }
#pragma unroll
  for (int m = 16; m >= 1; m >>= 1) { s += __shfl_xor(s, m, 32); s2 += __shfl_xor(s2, m, 32); }
  const int wv = threadIdx.x >> 5;
  if ((threadIdx.x & 31) == 0) { red0[wv] = s; red1[wv] = s2; }
  __syncthreads();
  float ts = 0.f, ts2 = 0.f;
#pragma unroll
  for (int i = 0; i < 8; ++i) { ts += red0[i]; ts2 += red1[i]; }
  const float mu = ts * (1.0f / Hh);
  const float var = ts2 * (1.0f / Hh) - mu * mu;
  const float rstd = rsqrtf(var + 1e-12f);
  for (int c = threadIdx.x; c < Hh; c += 256) {
    float v = (xr[c] - mu) * rstd * sc[c] + bi[c];
    if (out_f32) out_f32[(size_t)row * Hh + c] = v;
    else         out_bf16[(size_t)row * Hh + c] = f2bf(v);
  }
}

// ---------------------------------------------------------------------------
// GEMM: OUT = X(bf16,[N,H]) @ W(bf16,[H,H])^T + bias (+ residual)
// One wave per 32x32 output tile (2x2 register blocking -> 4 WMMAs reuse
// 2 A-frags + 2 B-frags). K-loop 1024 in chunks of 32 (bf16 WMMA).
// out_mode 0: f32 [N,H] (+residual)    1: bf16 [B,NH,S,DH]    2: bf16 [B,NH,DH,S]
// ---------------------------------------------------------------------------
__global__ __launch_bounds__(128) void k_gemm_proj(const unsigned short* __restrict__ X,
                                                   const unsigned short* __restrict__ W,
                                                   const float* __restrict__ bias,
                                                   const float* __restrict__ residual,
                                                   void* __restrict__ out, int out_mode) {
  const int wv = threadIdx.x >> 5;
  const int lane = threadIdx.x & 31;
  const int hf = lane >> 4, l15 = lane & 15;
  const int wt = blockIdx.x * 4 + wv;       // 4096 tiles: (N/32) x (H/32) = 128 x 32
  const int tileM = wt >> 5;                // 0..127
  const int tileN = wt & 31;                // 0..31
  const unsigned short* xr0 = X + (size_t)(tileM * 32 + l15) * Hh;
  const unsigned short* xr1 = X + (size_t)(tileM * 32 + 16 + l15) * Hh;
  const unsigned short* wr0 = W + (size_t)(tileN * 32 + l15) * Hh;
  const unsigned short* wr1 = W + (size_t)(tileN * 32 + 16 + l15) * Hh;
  v8f acc[2][2];
  acc[0][0] = vzero8(); acc[0][1] = vzero8();
  acc[1][0] = vzero8(); acc[1][1] = vzero8();
#pragma unroll 2
  for (int k0 = 0; k0 < Hh; k0 += 32) {
    v16bf a0 = ld_fragA(xr0 + k0 + hf * 16, xr0 + k0 + 8 + hf * 16);
    v16bf a1 = ld_fragA(xr1 + k0 + hf * 16, xr1 + k0 + 8 + hf * 16);
    v16bf b0 = ld_frag16(wr0 + k0 + hf * 16);
    v16bf b1 = ld_frag16(wr1 + k0 + hf * 16);
    acc[0][0] = wmma_bf16(a0, b0, acc[0][0]);
    acc[0][1] = wmma_bf16(a0, b1, acc[0][1]);
    acc[1][0] = wmma_bf16(a1, b0, acc[1][0]);
    acc[1][1] = wmma_bf16(a1, b1, acc[1][1]);
  }
#pragma unroll
  for (int ni = 0; ni < 2; ++ni) {
    const int col = tileN * 32 + ni * 16 + l15;
    const float bn = bias[col];
#pragma unroll
    for (int mi = 0; mi < 2; ++mi) {
      if (out_mode == 0) {
        float* o = (float*)out;
#pragma unroll
        for (int j = 0; j < 8; ++j) {
          int row = tileM * 32 + mi * 16 + j + hf * 8;
          size_t idx = (size_t)row * Hh + col;
          float v = acc[mi][ni][j] + bn;
          if (residual) v += residual[idx];
          o[idx] = v;
        }
      } else if (out_mode == 1) {             // q/k: [B,NH,S,DH]
        unsigned short* o = (unsigned short*)out;
        const int h_ = col >> 6, d_ = col & 63;
#pragma unroll
        for (int j = 0; j < 8; ++j) {
          int g = tileM * 32 + mi * 16 + j + hf * 8;
          int b_ = g >> 10, s_ = g & 1023;
          o[(((size_t)(b_ * NHh + h_) * Ss) + s_) * DHh + d_] = f2bf(acc[mi][ni][j] + bn);
        }
      } else {                                // v transposed: [B,NH,DH,S]
        unsigned short* o = (unsigned short*)out;
        const int h_ = col >> 6, d_ = col & 63;
#pragma unroll
        for (int j = 0; j < 8; ++j) {
          int g = tileM * 32 + mi * 16 + j + hf * 8;
          int b_ = g >> 10, s_ = g & 1023;
          o[(((size_t)(b_ * NHh + h_) * DHh) + d_) * Ss + s_] = f2bf(acc[mi][ni][j] + bn);
        }
      }
    }
  }
}

// ---------------------------------------------------------------------------
// Band dots: s01[i] = gq[i]·gk[i+1]/H ; s10[i] = gq[i+1]·gk[i]/H. One wave per (b,i).
// ---------------------------------------------------------------------------
__global__ __launch_bounds__(32) void k_band_dots(const float* __restrict__ gq,
                                                  const float* __restrict__ gk,
                                                  float* __restrict__ s01,
                                                  float* __restrict__ s10) {
  const int i = blockIdx.x % (Ss - 1);
  const int b = blockIdx.x / (Ss - 1);
  const float* q0 = gq + ((size_t)b * Ss + i) * Hh;
  const float* q1 = q0 + Hh;
  const float* k0 = gk + ((size_t)b * Ss + i) * Hh;
  const float* k1 = k0 + Hh;
  float a = 0.f, c = 0.f;
  for (int d = threadIdx.x; d < Hh; d += 32) { a += q0[d] * k1[d]; c += q1[d] * k0[d]; }
#pragma unroll
  for (int m = 16; m >= 1; m >>= 1) { a += __shfl_xor(a, m, 32); c += __shfl_xor(c, m, 32); }
  if (threadIdx.x == 0) {
    s01[b * Ss + i] = a * (1.0f / Hh);
    s10[b * Ss + i] = c * (1.0f / Hh);
  }
}

// ---------------------------------------------------------------------------
// Band softmax + symmetrization + prior mix + log prefix-sum. One block per batch.
// ---------------------------------------------------------------------------
__global__ __launch_bounds__(32) void k_band_scan(const float* __restrict__ s01,
                                                  const float* __restrict__ s10,
                                                  const int* __restrict__ am,
                                                  const float* __restrict__ gp,
                                                  float* __restrict__ pr, float* __restrict__ pl,
                                                  float* __restrict__ band, float* __restrict__ Cc) {
  if (threadIdx.x != 0) return;
  const int b = blockIdx.x;
  // per-row 2-entry masked softmax (nan_to_num semantics)
  for (int i = 0; i < Ss; ++i) {
    float vl = -3.0e38f, vr = -3.0e38f;
    if (i > 0 && am[((size_t)b * Ss + i) * Ss + (i - 1)] != 0) vl = s10[b * Ss + i - 1];
    if (i < Ss - 1 && am[((size_t)b * Ss + i) * Ss + (i + 1)] != 0) vr = s01[b * Ss + i];
    float mx = fmaxf(vl, vr);
    float el = 0.f, er = 0.f;
    if (mx > -1.0e37f) {
      if (vl > -1.0e37f) el = expf(vl - mx);
      if (vr > -1.0e37f) er = expf(vr - mx);
    }
    float den = el + er;
    pr[b * Ss + i] = (den > 0.f) ? er / den : 0.f;
    pl[b * Ss + i] = (den > 0.f) ? el / den : 0.f;
  }
  // symmetrized band + prior mixing on superdiagonal + cumulative log
  float c = 0.f;
  Cc[b * Ss + 0] = 0.f;
  for (int i = 0; i < Ss - 1; ++i) {
    float braw = sqrtf(pr[b * Ss + i] * pl[b * Ss + i + 1] + 1e-9f);
    band[b * Ss + i] = braw;
    float g = gp[((size_t)b * Ss + i) * Ss + (i + 1)];
    float asup = g + (1.0f - g) * braw;
    c += logf(asup + 1e-9f);
    Cc[b * Ss + i + 1] = c;
  }
  band[b * Ss + Ss - 1] = 0.f;
}

// ---------------------------------------------------------------------------
// Fill g_attn and break_prob outputs elementwise. One thread per element.
// ---------------------------------------------------------------------------
__global__ __launch_bounds__(256) void k_write_gmats(const float* __restrict__ gp,
                                                     const float* __restrict__ band,
                                                     const float* __restrict__ Cc,
                                                     float* __restrict__ g_out,
                                                     float* __restrict__ bp_out) {
  const size_t idx = (size_t)blockIdx.x * 256 + threadIdx.x;   // B*S*S elements exactly
  const int b = (int)(idx >> 20);
  const int rem = (int)(idx & ((1u << 20) - 1));
  const int i = rem >> 10, j = rem & 1023;
  const float SQ = 3.16227766e-5f;    // sqrt(1e-9)
  float nraw = SQ;
  if (j == i + 1) nraw = band[b * Ss + i];
  else if (i == j + 1) nraw = band[b * Ss + j];
  const float g = gp[idx];
  const float bp = g + (1.0f - g) * nraw;
  bp_out[idx] = bp;
  float gv;
  if (i == j) gv = bp;
  else {
    int mx = (i > j) ? i : j, mn = (i > j) ? j : i;
    gv = expf(Cc[b * Ss + mx] - Cc[b * Ss + mn]) + 1e-9f;
  }
  g_out[idx] = gv;
}

// ---------------------------------------------------------------------------
// Flash attention: one wave per (b,h,16 q-rows). bf16 WMMA for qk^T and p@v,
// online softmax with shfl row-reductions, g_attn reweighting from prefix sums,
// masked scores streamed to d_out, ctx written bf16 [B,S,H].
// ---------------------------------------------------------------------------
__global__ __launch_bounds__(128) void k_flash_attn(const unsigned short* __restrict__ qm,
                                                    const unsigned short* __restrict__ km,
                                                    const unsigned short* __restrict__ vtm,
                                                    const int* __restrict__ am,
                                                    const float* __restrict__ Cc,
                                                    const float* __restrict__ g_full,
                                                    float* __restrict__ scores_out,
                                                    unsigned short* __restrict__ ctx_out) {
  __shared__ unsigned short pbuf[4][16 * 32];
  const int wv = threadIdx.x >> 5;
  const int lane = threadIdx.x & 31;
  const int hf = lane >> 4, l15 = lane & 15;
  const int wid = blockIdx.x * 4 + wv;      // 4096 waves = B*NH*(S/16)
  const int qblk = wid & 63;
  const int bh = wid >> 6;                  // b*NH + h
  const int b = bh >> 4;
  const int h = bh & 15;
  const int qbase = qblk * 16;

  const unsigned short* qrow = qm + ((size_t)bh * Ss + qbase + l15) * DHh;
  const v16bf aq0 = ld_fragA(qrow + hf * 16, qrow + 8 + hf * 16);         // dh 0..31
  const v16bf aq1 = ld_fragA(qrow + 32 + hf * 16, qrow + 40 + hf * 16);    // dh 32..63

  float Cq[8], gdiag[8], mrow[8], lrow[8];
  v8f acc[4];
#pragma unroll
  for (int j = 0; j < 8; ++j) {
    int r = qbase + j + hf * 8;
    Cq[j] = Cc[b * Ss + r];
    gdiag[j] = g_full[((size_t)b * Ss + r) * Ss + r];
    mrow[j] = -3.0e38f;
    lrow[j] = 0.f;
  }
#pragma unroll
  for (int t = 0; t < 4; ++t) acc[t] = vzero8();

  for (int k0 = 0; k0 < Ss; k0 += 32) {
    // ---- score tile: 16 q rows x 32 keys (two 16x16 accumulators) ----
    v8f s[2];
#pragma unroll
    for (int t = 0; t < 2; ++t) {
      const unsigned short* kr = km + ((size_t)bh * Ss + k0 + t * 16 + l15) * DHh;
      v16bf kb0 = ld_frag16(kr + hf * 16);
      v16bf kb1 = ld_frag16(kr + 32 + hf * 16);
      s[t] = wmma_bf16(aq0, kb0, vzero8());
      s[t] = wmma_bf16(aq1, kb1, s[t]);
    }
    int colv[2];
    float Ck[2];
#pragma unroll
    for (int t = 0; t < 2; ++t) { colv[t] = k0 + t * 16 + l15; Ck[t] = Cc[b * Ss + colv[t]]; }
    // ---- scale + mask + stream scores to output ----
#pragma unroll
    for (int t = 0; t < 2; ++t) {
#pragma unroll
      for (int j = 0; j < 8; ++j) {
        int row = qbase + j + hf * 8;
        float sv = s[t][j] * 0.125f;   // 1/sqrt(DH)
        bool keep = (row == colv[t]) || (am[((size_t)b * Ss + row) * Ss + colv[t]] != 0);
        sv = keep ? sv : -__builtin_inff();
        scores_out[((size_t)bh * Ss + row) * Ss + colv[t]] = sv;
        s[t][j] = sv;
      }
    }
    // ---- online softmax stats (rows distributed as (VGPR j, lane-half)) ----
    float mnew[8], corr[8], rsum[8];
#pragma unroll
    for (int j = 0; j < 8; ++j) {
      float t0 = fmaxf(s[0][j], s[1][j]);
      t0 = fmaxf(t0, __shfl_xor(t0, 1, 32));
      t0 = fmaxf(t0, __shfl_xor(t0, 2, 32));
      t0 = fmaxf(t0, __shfl_xor(t0, 4, 32));
      t0 = fmaxf(t0, __shfl_xor(t0, 8, 32));
      mnew[j] = fmaxf(mrow[j], t0);
      corr[j] = (mnew[j] > -1.0e37f) ? expf(mrow[j] - mnew[j]) : 0.f;
      rsum[j] = 0.f;
    }
    // ---- probs (with g_attn reweight) -> LDS in bf16, C-layout tile ----
#pragma unroll
    for (int t = 0; t < 2; ++t) {
#pragma unroll
      for (int j = 0; j < 8; ++j) {
        int row = qbase + j + hf * 8;
        float pe = (s[t][j] > -1.0e37f) ? expf(s[t][j] - mnew[j]) : 0.f;
        rsum[j] += pe;
        float gv = (row == colv[t]) ? gdiag[j]
                 : (expf((colv[t] > row) ? (Ck[t] - Cq[j]) : (Cq[j] - Ck[t])) + 1e-9f);
        pbuf[wv][(j + hf * 8) * 32 + t * 16 + l15] = f2bf(pe * gv);
      }
    }
#pragma unroll
    for (int j = 0; j < 8; ++j) {
      float r = rsum[j];
      r += __shfl_xor(r, 1, 32);
      r += __shfl_xor(r, 2, 32);
      r += __shfl_xor(r, 4, 32);
      r += __shfl_xor(r, 8, 32);
      lrow[j] = lrow[j] * corr[j] + r;
      mrow[j] = mnew[j];
    }
#pragma unroll
    for (int t = 0; t < 4; ++t)
#pragma unroll
      for (int j = 0; j < 8; ++j) acc[t][j] *= corr[j];
    __syncthreads();
    // ---- probs as A-fragment (from LDS) x V (B-fragments from vT) ----
    const unsigned short* pw = &pbuf[wv][0];
    v16bf pa = ld_fragA(pw + l15 * 32 + hf * 16, pw + l15 * 32 + 8 + hf * 16);
#pragma unroll
    for (int t = 0; t < 4; ++t) {
      const unsigned short* vr = vtm + ((size_t)bh * DHh + t * 16 + l15) * Ss + k0 + hf * 16;
      acc[t] = wmma_bf16(pa, ld_frag16(vr), acc[t]);
    }
    __syncthreads();
  }
  // ---- normalize and write ctx bf16 [B,S,H] ----
  float inv[8];
#pragma unroll
  for (int j = 0; j < 8; ++j) inv[j] = 1.0f / lrow[j];
#pragma unroll
  for (int t = 0; t < 4; ++t) {
#pragma unroll
    for (int j = 0; j < 8; ++j) {
      int row = qbase + j + hf * 8;
      ctx_out[((size_t)b * Ss + row) * Hh + h * DHh + t * 16 + l15] = f2bf(acc[t][j] * inv[j]);
    }
  }
}

// ---------------------------------------------------------------------------
// Host-side orchestration
// ---------------------------------------------------------------------------
extern "C" void kernel_launch(void* const* d_in, const int* in_sizes, int n_in,
                              void* d_out, int out_size, void* d_ws, size_t ws_size,
                              hipStream_t stream) {
  (void)in_sizes; (void)n_in; (void)out_size; (void)ws_size;

  const float* hidden = (const float*)d_in[0];
  const float* gp     = (const float*)d_in[1];
  const int*   am     = (const int*)d_in[2];
  const float* Wq  = (const float*)d_in[3];  const float* bq  = (const float*)d_in[4];
  const float* Wk  = (const float*)d_in[5];  const float* bk  = (const float*)d_in[6];
  const float* Wv  = (const float*)d_in[7];  const float* bv  = (const float*)d_in[8];
  const float* gWq = (const float*)d_in[9];  const float* gbq = (const float*)d_in[10];
  const float* gWk = (const float*)d_in[11]; const float* gbk = (const float*)d_in[12];
  const float* gls = (const float*)d_in[13]; const float* glb = (const float*)d_in[14];
  const float* Wo  = (const float*)d_in[15]; const float* bo  = (const float*)d_in[16];
  const float* ols = (const float*)d_in[17]; const float* olb = (const float*)d_in[18];

  // d_out layout (f32): out | scores | g_attn | break_prob
  float* out_o    = (float*)d_out;
  float* scores_o = out_o + (size_t)Bb * Ss * Hh;
  float* g_o      = scores_o + (size_t)Bb * NHh * Ss * Ss;
  float* bp_o     = g_o + (size_t)Bb * Ss * Ss;

  // workspace layout (bytes)
  unsigned char* w = (unsigned char*)d_ws;
  unsigned short* x_bf   = (unsigned short*)(w + 0);
  unsigned short* hs_bf  = (unsigned short*)(w + 8388608ULL);
  unsigned short* gWq_bf = (unsigned short*)(w + 16777216ULL);
  unsigned short* gWk_bf = (unsigned short*)(w + 18874368ULL);
  unsigned short* Wq_bf  = (unsigned short*)(w + 20971520ULL);
  unsigned short* Wk_bf  = (unsigned short*)(w + 23068672ULL);
  unsigned short* Wv_bf  = (unsigned short*)(w + 25165824ULL);
  unsigned short* Wo_bf  = (unsigned short*)(w + 27262976ULL);
  unsigned short* q_bf   = (unsigned short*)(w + 29360128ULL);
  unsigned short* k_bf   = (unsigned short*)(w + 37748736ULL);
  unsigned short* vt_bf  = (unsigned short*)(w + 46137344ULL);
  unsigned short* ctx_bf = (unsigned short*)(w + 54525952ULL);
  float* gq_f  = (float*)(w + 62914560ULL);
  float* gk_f  = (float*)(w + 79691776ULL);
  float* tmp_f = gq_f;                       // reuse after band dots
  float* s01_f = (float*)(w + 96468992ULL);
  float* s10_f = (float*)(w + 96485376ULL);
  float* pr_f  = (float*)(w + 96501760ULL);
  float* pl_f  = (float*)(w + 96518144ULL);
  float* band_f= (float*)(w + 96534528ULL);
  float* C_f   = (float*)(w + 96550912ULL);

  const int NE  = Nrows * Hh;   // 4,194,304
  const int NW  = Hh * Hh;      // 1,048,576

  // 1) bf16 conversions
  k_cvt_bf16<<<(NE + 255) / 256, 256, 0, stream>>>(hidden, x_bf, NE);
  k_cvt_bf16<<<(NW + 255) / 256, 256, 0, stream>>>(gWq, gWq_bf, NW);
  k_cvt_bf16<<<(NW + 255) / 256, 256, 0, stream>>>(gWk, gWk_bf, NW);
  k_cvt_bf16<<<(NW + 255) / 256, 256, 0, stream>>>(Wq, Wq_bf, NW);
  k_cvt_bf16<<<(NW + 255) / 256, 256, 0, stream>>>(Wk, Wk_bf, NW);
  k_cvt_bf16<<<(NW + 255) / 256, 256, 0, stream>>>(Wv, Wv_bf, NW);
  k_cvt_bf16<<<(NW + 255) / 256, 256, 0, stream>>>(Wo, Wo_bf, NW);

  // 2) group LayerNorm -> hs (bf16)
  k_layernorm<<<Nrows, 256, 0, stream>>>(hidden, gls, glb, nullptr, hs_bf);

  // 3) projections (WMMA bf16 GEMMs, 32x32 tile per wave)
  k_gemm_proj<<<1024, 128, 0, stream>>>(hs_bf, gWq_bf, gbq, nullptr, gq_f, 0);
  k_gemm_proj<<<1024, 128, 0, stream>>>(hs_bf, gWk_bf, gbk, nullptr, gk_f, 0);
  k_gemm_proj<<<1024, 128, 0, stream>>>(x_bf, Wq_bf, bq, nullptr, q_bf, 1);
  k_gemm_proj<<<1024, 128, 0, stream>>>(x_bf, Wk_bf, bk, nullptr, k_bf, 1);
  k_gemm_proj<<<1024, 128, 0, stream>>>(x_bf, Wv_bf, bv, nullptr, vt_bf, 2);

  // 4) group-attention band -> prefix sums
  k_band_dots<<<Bb * (Ss - 1), 32, 0, stream>>>(gq_f, gk_f, s01_f, s10_f);
  k_band_scan<<<Bb, 32, 0, stream>>>(s01_f, s10_f, am, gp, pr_f, pl_f, band_f, C_f);

  // 5) g_attn / break_prob outputs
  k_write_gmats<<<(Bb * Ss * Ss) / 256, 256, 0, stream>>>(gp, band_f, C_f, g_o, bp_o);

  // 6) flash attention (scores out + ctx bf16)
  k_flash_attn<<<1024, 128, 0, stream>>>(q_bf, k_bf, vt_bf, am, C_f, g_o, scores_o, ctx_bf);

  // 7) output projection + residual, then LayerNorm -> out
  k_gemm_proj<<<1024, 128, 0, stream>>>(ctx_bf, Wo_bf, bo, hidden, tmp_f, 0);
  k_layernorm<<<Nrows, 256, 0, stream>>>(tmp_f, ols, olb, out_o, nullptr);
}
